// AffinityMah_51118700757139
// MI455X (gfx1250) — matrix operand order repaired and checked by reading, not compiled, not run on hardware
//
#include <hip/hip_runtime.h>

// ---------------------------------------------------------------------------
// AffinityMah for MI455X (gfx1250, wave32, WMMA)
//   quad[b,n,m] = relu( ||Y[b,n]A||^2 + ||X[b,m]A||^2 - 2*(Y[b,n]A)·(X[b,m]A) )
// Pipeline:
//   prepA :  A (256x100 f32, row-major) -> At (128x256 f16, column-major, padded)
//   gemm1 :  XA/YA = rows x At via v_wmma_f32_16x16x32_f16; store f16 rows
//            (b*2048+row, 128) + fp32 sq-norms of the *rounded* f16 values
//   cross :  per batch: YA(16x32) x XA^T(32x16) WMMA tiles, 2-deep software
//            pipeline, fused relu epilogue, non-temporal fp32 stores
//            (output-write bound: 134 MB @ 23.3 TB/s ~ 6 us floor)
// Workspace layout (bytes, all 256-aligned; total ~8.2 MB):
//   [0)           At   : 128*256*2          = 65536
//   [65536)       XAf  : 16384*128*2        = 4194304
//   [4259840)     YAf  : 16384*128*2        = 4194304
//   [8454144)     sqX  : 16384*4            = 65536
//   [8519680)     sqY  : 16384*4            = 65536
// ---------------------------------------------------------------------------

typedef __attribute__((ext_vector_type(16))) _Float16 v16h;
typedef __attribute__((ext_vector_type(8)))  _Float16 v8h;
typedef __attribute__((ext_vector_type(8)))  float    v8f;

#define BATCH 8
#define MXDIM 2048
#define NYDIM 2048
#define DDIM  256
#define KDIM  100
#define KP    128

__device__ __forceinline__ v16h concat8(v8h lo, v8h hi) {
  return __builtin_shufflevector(lo, hi, 0,1,2,3,4,5,6,7,8,9,10,11,12,13,14,15);
}

// --------------------------- kernel 1: pad/transpose A ---------------------
// At[kcol][d] = (kcol < 100) ? (f16)A[d*100 + kcol] : 0    (At: 128 x 256 f16)
__global__ void prepA_kernel(const float* __restrict__ A, _Float16* __restrict__ At) {
  int idx  = blockIdx.x * 256 + threadIdx.x;   // 0 .. 32767
  int kcol = idx >> 8;
  int d    = idx & 255;
  float v  = (kcol < KDIM) ? A[d * KDIM + kcol] : 0.0f;
  At[idx]  = (_Float16)v;                      // idx == kcol*256 + d
}

// ------------------ kernel 2: XA / YA projection + sq norms ----------------
// 256 threads = 8 waves; each wave owns a 16-row x 128-col output tile.
// Row id r in [0, 32768): r < 16384 -> X row r, else Y row r-16384.
__global__ void gemm1_kernel(const float* __restrict__ X, const float* __restrict__ Y,
                             const _Float16* __restrict__ At,
                             _Float16* __restrict__ XAf, _Float16* __restrict__ YAf,
                             float* __restrict__ sqX, float* __restrict__ sqY) {
  __shared__ _Float16 tile[8][16][136];   // 136-halfword row stride: bank-spread
  __shared__ float    sqp[8][2][16];

  const int wave = threadIdx.x >> 5;
  const int lane = threadIdx.x & 31;
  const int lr   = lane & 15;             // row within 16-row tile / matrix col
  const int hi   = lane >> 4;             // half-wave select

  const int rowGlobal  = blockIdx.x * 128 + wave * 16;
  const bool isY       = rowGlobal >= (BATCH * MXDIM);
  const int  srcBase   = isY ? rowGlobal - BATCH * MXDIM : rowGlobal;
  const float*    src  = isY ? Y   : X;
  _Float16*       dstF = isY ? YAf : XAf;
  float*          dstS = isY ? sqY : sqX;

  const float* rp = src + (size_t)(srcBase + lr) * DDIM;

  v8f acc[8];
#pragma unroll
  for (int n = 0; n < 8; ++n) acc[n] = 0.0f;

  // Double-buffered f32 row chunk (HBM-latency loads prefetched one chunk ahead)
  float4 xq[2][4];
  {
    const float* p = rp + hi * 8;
    xq[0][0] = *(const float4*)(p);
    xq[0][1] = *(const float4*)(p + 4);
    xq[0][2] = *(const float4*)(p + 16);
    xq[0][3] = *(const float4*)(p + 20);
  }

#pragma unroll
  for (int dc = 0; dc < 8; ++dc) {        // K = 256 in chunks of 32
    const int cur = dc & 1, nxt = cur ^ 1;
    const int d0  = dc * 32;
    if (dc < 7) {                         // prefetch next f32 chunk
      const float* p = rp + d0 + 32 + hi * 8;
      xq[nxt][0] = *(const float4*)(p);
      xq[nxt][1] = *(const float4*)(p + 4);
      xq[nxt][2] = *(const float4*)(p + 16);
      xq[nxt][3] = *(const float4*)(p + 20);
    }
    // All 8 B fragments issued as one load group (L2-hot; waits then drain
    // gradually instead of s_wait_loadcnt 0 before every WMMA)
    v16h bfar[8];
#pragma unroll
    for (int n = 0; n < 8; ++n) {
      const _Float16* bp = At + (size_t)(n * 16 + lr) * DDIM + d0 + hi * 8;
      bfar[n] = concat8(*(const v8h*)bp, *(const v8h*)(bp + 16));
    }
    __builtin_amdgcn_sched_barrier(0);    // keep load group ahead of WMMA chain

    const float4 x0 = xq[cur][0], x1 = xq[cur][1], x2 = xq[cur][2], x3 = xq[cur][3];
    v16h af;
    af[0]=(_Float16)x0.x;  af[1]=(_Float16)x0.y;  af[2]=(_Float16)x0.z;  af[3]=(_Float16)x0.w;
    af[4]=(_Float16)x1.x;  af[5]=(_Float16)x1.y;  af[6]=(_Float16)x1.z;  af[7]=(_Float16)x1.w;
    af[8]=(_Float16)x2.x;  af[9]=(_Float16)x2.y;  af[10]=(_Float16)x2.z; af[11]=(_Float16)x2.w;
    af[12]=(_Float16)x3.x; af[13]=(_Float16)x3.y; af[14]=(_Float16)x3.z; af[15]=(_Float16)x3.w;
#pragma unroll
    for (int n = 0; n < 8; ++n)
      acc[n] = __builtin_amdgcn_wmma_f32_16x16x32_f16(
          false, af, false, bfar[n], (short)0, acc[n], false, false);
  }

  // D layout: lane lr = column (within n-subtile), VGPR v -> row v + hi*8.
  // Round to f16 into LDS, then emit coalesced row stores + fp32 norms of the
  // rounded values (keeps quad = sqY+sqX-2*cross self-consistent).
#pragma unroll
  for (int n = 0; n < 8; ++n) {
#pragma unroll
    for (int v = 0; v < 8; ++v)
      tile[wave][v + hi * 8][n * 16 + lr] = (_Float16)acc[n][v];
  }
  __syncthreads();

  float s = 0.0f;
  const _Float16* rowp = &tile[wave][lr][hi * 64];
  _Float16*       gp   = dstF + (size_t)(srcBase + lr) * KP + hi * 64;
#pragma unroll
  for (int c = 0; c < 64; c += 8) {
    v8h h = *(const v8h*)(rowp + c);
    *(v8h*)(gp + c) = h;
#pragma unroll
    for (int e = 0; e < 8; ++e) { float f = (float)h[e]; s += f * f; }
  }
  sqp[wave][hi][lr] = s;
  __syncthreads();
  if (hi == 0) dstS[srcBase + lr] = sqp[wave][0][lr] + sqp[wave][1][lr];
}

// --------------------- kernel 3: cross GEMM + fused epilogue ---------------
// grid (MX/128, NY/128, B); 8 waves per block; wave tile = 32(n) x 64(m).
__device__ __forceinline__ void load_chunk(const _Float16* __restrict__ Yb,
                                           const _Float16* __restrict__ Xb,
                                           int nb, int mb, int lr, int hi, int d0,
                                           v16h (&a)[2], v16h (&bf)[4]) {
#pragma unroll
  for (int i = 0; i < 2; ++i) {           // A layout: split-K {0..7, 16..23}
    const _Float16* p = Yb + (size_t)(nb + i * 16 + lr) * KP + d0 + hi * 8;
    a[i] = concat8(*(const v8h*)p, *(const v8h*)(p + 16));
  }
#pragma unroll
  for (int j = 0; j < 4; ++j) {           // B layout: split-K {0..7, 16..23}
    const _Float16* p = Xb + (size_t)(mb + j * 16 + lr) * KP + d0 + hi * 8;
    bf[j] = concat8(*(const v8h*)p, *(const v8h*)(p + 16));
  }
}

__global__ void cross_kernel(const _Float16* __restrict__ XAf,
                             const _Float16* __restrict__ YAf,
                             const float* __restrict__ sqX,
                             const float* __restrict__ sqY,
                             float* __restrict__ out) {
  const int wave = threadIdx.x >> 5;
  const int lane = threadIdx.x & 31;
  const int lr   = lane & 15;
  const int hi   = lane >> 4;
  const int b    = blockIdx.z;
  const int nb   = blockIdx.y * 128 + (wave >> 1) * 32;
  const int mb   = blockIdx.x * 128 + (wave & 1) * 64;

  const _Float16* Yb = YAf + (size_t)b * NYDIM * KP;
  const _Float16* Xb = XAf + (size_t)b * MXDIM * KP;

  // Epilogue scalars issued first: trivial loads, long time to retire.
  const float* sqXb = sqX + b * MXDIM;
  const float* sqYb = sqY + b * NYDIM;
  float sy[2][8], sx[4];
#pragma unroll
  for (int i = 0; i < 2; ++i)
#pragma unroll
    for (int v = 0; v < 8; ++v)
      sy[i][v] = sqYb[nb + i * 16 + hi * 8 + v];
#pragma unroll
  for (int j = 0; j < 4; ++j) sx[j] = sqXb[mb + j * 16 + lr];

  v8f acc[2][4];
#pragma unroll
  for (int i = 0; i < 2; ++i)
#pragma unroll
    for (int j = 0; j < 4; ++j) acc[i][j] = 0.0f;

  // 2-deep software pipeline over K = 128 (4 chunks of 32, all L2-hot):
  // chunk dc+1's 12 b128 loads are in flight while chunk dc's 8 WMMAs run.
  v16h aA[2][2], bB[2][4];
  load_chunk(Yb, Xb, nb, mb, lr, hi, 0, aA[0], bB[0]);
#pragma unroll
  for (int dc = 0; dc < 4; ++dc) {
    const int cur = dc & 1, nxt = cur ^ 1;
    if (dc < 3)
      load_chunk(Yb, Xb, nb, mb, lr, hi, (dc + 1) * 32, aA[nxt], bB[nxt]);
    __builtin_amdgcn_sched_barrier(0);    // loads stay ahead of the WMMA chain
#pragma unroll
    for (int i = 0; i < 2; ++i)
#pragma unroll
      for (int j = 0; j < 4; ++j)
        acc[i][j] = __builtin_amdgcn_wmma_f32_16x16x32_f16(
            false, aA[cur][i], false, bB[cur][j], (short)0, acc[i][j], false, false);
  }

#pragma unroll
  for (int j = 0; j < 4; ++j) {
    const int m = mb + j * 16 + lr;       // D layout: lane = m column
#pragma unroll
    for (int i = 0; i < 2; ++i) {
      const int n0 = nb + i * 16 + hi * 8;
#pragma unroll
      for (int v = 0; v < 8; ++v) {       // VGPR v -> n row
        float q = sy[i][v] + sx[j] - 2.0f * acc[i][j][v];
        q = q > 0.0f ? q : 0.0f;
        __builtin_nontemporal_store(q, out + ((size_t)b * NYDIM + (n0 + v)) * MXDIM + m);
      }
    }
  }
}

// ---------------------------------------------------------------------------
extern "C" void kernel_launch(void* const* d_in, const int* in_sizes, int n_in,
                              void* d_out, int out_size, void* d_ws, size_t ws_size,
                              hipStream_t stream) {
  (void)in_sizes; (void)n_in; (void)out_size; (void)ws_size;
  const float* X = (const float*)d_in[0];   // (8, 2048, 256) f32
  const float* Y = (const float*)d_in[1];   // (8, 2048, 256) f32
  const float* A = (const float*)d_in[2];   // (256, 100)     f32
  float* out = (float*)d_out;               // (8, 2048, 2048) f32

  char* ws = (char*)d_ws;
  _Float16* At  = (_Float16*)(ws);
  _Float16* XAf = (_Float16*)(ws + 65536);
  _Float16* YAf = (_Float16*)(ws + 65536 + 4194304);
  float*    sqX = (float*)   (ws + 65536 + 2 * 4194304);
  float*    sqY = (float*)   (ws + 65536 + 2 * 4194304 + 65536);

  prepA_kernel<<<128, 256, 0, stream>>>(A, At);
  gemm1_kernel<<<256, 256, 0, stream>>>(X, Y, At, XAf, YAf, sqX, sqY);
  dim3 grid(MXDIM / 128, NYDIM / 128, BATCH);
  cross_kernel<<<grid, 256, 0, stream>>>(XAf, YAf, sqX, sqY, out);
}
